// DAHHConv_dis_11639361372553
// MI455X (gfx1250) — matrix operand order repaired and checked by reading, not compiled, not run on hardware
//
#include <hip/hip_runtime.h>
#include <stdint.h>

// ---------------- problem constants (from reference) ----------------
#define B_   4
#define N_   4096
#define E_   2048
#define C_   64      // C_IN == C_OUT == 64

// ---------------- CDNA5 WMMA types ----------------
typedef __attribute__((ext_vector_type(16))) __bf16 v16bf;
typedef __attribute__((ext_vector_type(8)))  float  v8f;

#define LDS_STRIDE 80   // elements; multiple of 8 (16B-aligned b128 ops), not pow2

__device__ __forceinline__ unsigned short f2bf(float f) {
  union { float f; unsigned u; } v; v.f = f;
  unsigned u = v.u;
  u += 0x7FFFu + ((u >> 16) & 1u);     // round-to-nearest-even
  return (unsigned short)(u >> 16);
}
// rounded pack (general floats: xt / he epilogues)
__device__ __forceinline__ unsigned pk2(float a, float b) {
  return (unsigned)f2bf(a) | ((unsigned)f2bf(b) << 16);
}
// truncating pack: ONE v_perm_b32. Bit-exact for H's {0.0, 1.0} values.
// result[15:0] = a[31:16], result[31:16] = b[31:16]
__device__ __forceinline__ unsigned pk2t(float a, float b) {
  return __builtin_amdgcn_perm(__float_as_uint(a), __float_as_uint(b), 0x03020706u);
}

// CDNA5 async global->LDS copy, 16 bytes per lane (ASYNCcnt-tracked)
__device__ __forceinline__ void async_copy16(const void* gsrc, void* lds_dst) {
  unsigned ldsa = (unsigned)(unsigned long long)lds_dst;  // low 32b = LDS offset
  asm volatile("global_load_async_to_lds_b128 %0, %1, off"
               :: "v"(ldsa), "v"(gsrc) : "memory");
}
__device__ __forceinline__ void async_wait0() {
  asm volatile("s_wait_asynccnt 0" ::: "memory");
}

// ======================================================================
// Kernel 0: xtT[b,c,n] = bf16( sum_k x[b,n,k] * theta[k,c] )  (f32 accum)
// Transposed output so GEMM1's B-tile staging is a contiguous async copy.
// ======================================================================
__global__ __launch_bounds__(256)
void hg_node_transform(const float* __restrict__ x,
                       const float* __restrict__ theta,
                       unsigned short* __restrict__ xtT) {
  __shared__ float th[64 * 64];
  __shared__ float xs[64 * 64];
  int t = threadIdx.x;
  int b  = blockIdx.x / (N_ / 64);
  int n0 = (blockIdx.x % (N_ / 64)) * 64;
  {
    const float4* tp  = (const float4*)theta;
    const float4* xp  = (const float4*)(x + ((size_t)b * N_ + n0) * C_);
    float4* thp = (float4*)th;
    float4* xsp = (float4*)xs;
    #pragma unroll
    for (int r = 0; r < 4; ++r) { thp[t + 256 * r] = tp[t + 256 * r];
                                  xsp[t + 256 * r] = xp[t + 256 * r]; }
  }
  __syncthreads();
  int c = t & 63, g = t >> 6;
  float acc[16];
  #pragma unroll
  for (int r = 0; r < 16; ++r) acc[r] = 0.f;
  for (int k = 0; k < 64; ++k) {
    float tv = th[k * 64 + c];
    #pragma unroll
    for (int r = 0; r < 16; ++r)
      acc[r] = fmaf(xs[(g * 16 + r) * 64 + k], tv, acc[r]);
  }
  uint4 w0, w1;
  w0.x = pk2(acc[0], acc[1]);   w0.y = pk2(acc[2], acc[3]);
  w0.z = pk2(acc[4], acc[5]);   w0.w = pk2(acc[6], acc[7]);
  w1.x = pk2(acc[8], acc[9]);   w1.y = pk2(acc[10], acc[11]);
  w1.z = pk2(acc[12], acc[13]); w1.w = pk2(acc[14], acc[15]);
  uint4* dst = (uint4*)&xtT[((size_t)b * C_ + c) * N_ + n0 + g * 16];
  dst[0] = w0; dst[1] = w1;
}

// ======================================================================
// Kernel 1 (edge aggregation):
//   heT[b,c,e] = (1/deg_e[b,e]) * sum_n H[b,n,e] * xt[b,n,c]
// One WG per (b, 64-wide e-tile); sweeps all N => deg_e fused.
// ======================================================================
__global__ __launch_bounds__(256)
void hg_edge_gemm(const float* __restrict__ H,
                  const unsigned short* __restrict__ xtT,
                  unsigned short* __restrict__ heT) {
  int b  = blockIdx.x / (E_ / 64);
  int e0 = (blockIdx.x % (E_ / 64)) * 64;
  const float*          Hb = H   + (size_t)b * N_ * E_;
  const unsigned short* xb = xtT + (size_t)b * C_ * N_;

  __shared__ unsigned short As[64 * LDS_STRIDE];   // As[e_local][k=n_local]
  __shared__ unsigned short Bs[64 * LDS_STRIDE];   // Bs[c][k=n_local]
  __shared__ float degred[256];
  __shared__ float dinv[64];

  int t = threadIdx.x, lane = t & 31, wave = t >> 5;
  int mt = wave & 3, ntbase = (wave >> 2) * 2;
  int cn = lane & 15, h8 = lane >> 4;
  int jcol = t & 63, ibase = t >> 6;    // A staging: column e, 16-row n block
  int brow = t >> 2, bseg = t & 3;      // B staging: row c, 16B segment

  v8f acc0 = {}, acc1 = {};
  float dacc = 0.f;

  for (int n0 = 0; n0 < N_; n0 += 64) {
    __syncthreads();
    // --- B tile: async global->LDS (already bf16, contiguous in k)
    {
      const unsigned short* src = xb + (size_t)brow * N_ + n0 + bseg * 8;
      unsigned short*       dst = &Bs[brow * LDS_STRIDE + bseg * 8];
      async_copy16(src, dst);
      async_copy16(src + 32, dst + 32);
    }
    // --- A tile: 16 coalesced b32 loads -> perm-pack -> 2 x ds_store_b128
    {
      const float* hp = &Hb[(size_t)(n0 + ibase * 16) * E_ + e0 + jcol];
      if (n0 + 64 < N_) __builtin_prefetch(hp + (size_t)64 * E_, 0, 0);
      float hv[16];
      #pragma unroll
      for (int r = 0; r < 16; ++r) hv[r] = hp[(size_t)r * E_];
      #pragma unroll
      for (int r = 0; r < 16; ++r) dacc += hv[r];
      uint4 w0, w1;
      w0.x = pk2t(hv[0], hv[1]);   w0.y = pk2t(hv[2], hv[3]);
      w0.z = pk2t(hv[4], hv[5]);   w0.w = pk2t(hv[6], hv[7]);
      w1.x = pk2t(hv[8], hv[9]);   w1.y = pk2t(hv[10], hv[11]);
      w1.z = pk2t(hv[12], hv[13]); w1.w = pk2t(hv[14], hv[15]);
      uint4* ap = (uint4*)&As[jcol * LDS_STRIDE + ibase * 16];
      ap[0] = w0; ap[1] = w1;
    }
    async_wait0();
    __syncthreads();
    // --- 2 WMMA K-steps of 32
    #pragma unroll
    for (int kk = 0; kk < 64; kk += 32) {
      int am = mt * 16 + cn;
      union { uint4 u[2]; v16bf v; } af, bf0, bf1;
      af.u[0] = *(const uint4*)(&As[am * LDS_STRIDE + kk + 16 * h8]);
      af.u[1] = *(const uint4*)(&As[am * LDS_STRIDE + kk + 16 * h8 + 8]);
      int bn0 = (ntbase + 0) * 16 + cn;
      int bn1 = (ntbase + 1) * 16 + cn;
      bf0.u[0] = *(const uint4*)(&Bs[bn0 * LDS_STRIDE + kk + 16 * h8]);
      bf0.u[1] = *(const uint4*)(&Bs[bn0 * LDS_STRIDE + kk + 16 * h8 + 8]);
      bf1.u[0] = *(const uint4*)(&Bs[bn1 * LDS_STRIDE + kk + 16 * h8]);
      bf1.u[1] = *(const uint4*)(&Bs[bn1 * LDS_STRIDE + kk + 16 * h8 + 8]);
      acc0 = __builtin_amdgcn_wmma_f32_16x16x32_bf16(false, af.v, false, bf0.v,
                                                     (short)0, acc0, false, false);
      acc1 = __builtin_amdgcn_wmma_f32_16x16x32_bf16(false, af.v, false, bf1.v,
                                                     (short)0, acc1, false, false);
    }
  }

  // --- fused deg_e reduction -> dinv
  degred[t] = dacc;              // unique (ibase, jcol) per thread
  __syncthreads();
  if (t < 64) {
    float d = degred[t] + degred[64 + t] + degred[128 + t] + degred[192 + t];
    dinv[t] = (d > 0.f) ? (1.0f / d) : 0.0f;
  }
  __syncthreads();

  // --- epilogue: heT[c][e], one b128 store per tile
  unsigned short* hb = heT + (size_t)b * C_ * E_;
  int mbase = mt * 16 + 8 * h8;
  #pragma unroll
  for (int p = 0; p < 2; ++p) {
    v8f a = p ? acc1 : acc0;
    int c = (ntbase + p) * 16 + cn;
    uint4 q;
    q.x = pk2(a[0] * dinv[mbase + 0], a[1] * dinv[mbase + 1]);
    q.y = pk2(a[2] * dinv[mbase + 2], a[3] * dinv[mbase + 3]);
    q.z = pk2(a[4] * dinv[mbase + 4], a[5] * dinv[mbase + 5]);
    q.w = pk2(a[6] * dinv[mbase + 6], a[7] * dinv[mbase + 7]);
    *(uint4*)&hb[(size_t)c * E_ + e0 + mbase] = q;
  }
}

// ======================================================================
// Kernel 2 (node aggregation):
//   out[b,n,c] = bias[c] + (1/deg_n[b,n]) * sum_e H[b,n,e] * he[b,e,c]
// One WG per (b, 64-row n-tile); sweeps all E => deg_n fused.
// ======================================================================
__global__ __launch_bounds__(256)
void hg_node_gemm(const float* __restrict__ H,
                  const unsigned short* __restrict__ heT,
                  const float* __restrict__ bias,
                  float* __restrict__ out) {
  int b  = blockIdx.x / (N_ / 64);
  int n0 = (blockIdx.x % (N_ / 64)) * 64;
  const float*          Hb = H   + (size_t)b * N_ * E_;
  const unsigned short* hb = heT + (size_t)b * C_ * E_;

  __shared__ unsigned short As[64 * LDS_STRIDE];   // As[n_local][k=e_local]
  __shared__ unsigned short Bs[64 * LDS_STRIDE];   // Bs[c][k=e_local]
  __shared__ float degred[256];
  __shared__ float dinv[64];

  int t = threadIdx.x, lane = t & 31, wave = t >> 5;
  int mt = wave & 3, ntbase = (wave >> 2) * 2;
  int cn = lane & 15, h8 = lane >> 4;
  int arow = t >> 2, aseg = t & 3;      // A staging: row n, 16-elem e chunk

  v8f acc0 = {}, acc1 = {};
  float dacc = 0.f;                     // partial of deg_n[n0 + arow]

  for (int e1 = 0; e1 < E_; e1 += 64) {
    __syncthreads();
    // --- B tile: async global->LDS from heT (bf16, contiguous in k)
    {
      const unsigned short* src = hb + (size_t)arow * E_ + e1 + aseg * 8;
      unsigned short*       dst = &Bs[arow * LDS_STRIDE + aseg * 8];
      async_copy16(src, dst);
      async_copy16(src + 32, dst + 32);
    }
    // --- A tile: 4 x global_load_b128 -> perm-pack -> 2 x ds_store_b128
    {
      const float4* hp4 = (const float4*)&Hb[(size_t)(n0 + arow) * E_ + e1 + aseg * 16];
      if (e1 + 64 < E_) __builtin_prefetch(hp4 + 16, 0, 0);
      float4 f0 = hp4[0], f1 = hp4[1], f2 = hp4[2], f3 = hp4[3];
      dacc += (f0.x + f0.y + f0.z + f0.w) + (f1.x + f1.y + f1.z + f1.w)
            + (f2.x + f2.y + f2.z + f2.w) + (f3.x + f3.y + f3.z + f3.w);
      uint4 w0, w1;
      w0.x = pk2t(f0.x, f0.y); w0.y = pk2t(f0.z, f0.w);
      w0.z = pk2t(f1.x, f1.y); w0.w = pk2t(f1.z, f1.w);
      w1.x = pk2t(f2.x, f2.y); w1.y = pk2t(f2.z, f2.w);
      w1.z = pk2t(f3.x, f3.y); w1.w = pk2t(f3.z, f3.w);
      uint4* ap = (uint4*)&As[arow * LDS_STRIDE + aseg * 16];
      ap[0] = w0; ap[1] = w1;
    }
    async_wait0();
    __syncthreads();
    #pragma unroll
    for (int kk = 0; kk < 64; kk += 32) {
      int am = mt * 16 + cn;
      union { uint4 u[2]; v16bf v; } af, bf0, bf1;
      af.u[0] = *(const uint4*)(&As[am * LDS_STRIDE + kk + 16 * h8]);
      af.u[1] = *(const uint4*)(&As[am * LDS_STRIDE + kk + 16 * h8 + 8]);
      int bn0 = (ntbase + 0) * 16 + cn;
      int bn1 = (ntbase + 1) * 16 + cn;
      bf0.u[0] = *(const uint4*)(&Bs[bn0 * LDS_STRIDE + kk + 16 * h8]);
      bf0.u[1] = *(const uint4*)(&Bs[bn0 * LDS_STRIDE + kk + 16 * h8 + 8]);
      bf1.u[0] = *(const uint4*)(&Bs[bn1 * LDS_STRIDE + kk + 16 * h8]);
      bf1.u[1] = *(const uint4*)(&Bs[bn1 * LDS_STRIDE + kk + 16 * h8 + 8]);
      acc0 = __builtin_amdgcn_wmma_f32_16x16x32_bf16(false, af.v, false, bf0.v,
                                                     (short)0, acc0, false, false);
      acc1 = __builtin_amdgcn_wmma_f32_16x16x32_bf16(false, af.v, false, bf1.v,
                                                     (short)0, acc1, false, false);
    }
  }

  // --- fused deg_n reduction (thread t = arow*4 + aseg)
  degred[t] = dacc;
  __syncthreads();
  if (t < 64) {
    float d = degred[4 * t] + degred[4 * t + 1] + degred[4 * t + 2] + degred[4 * t + 3];
    dinv[t] = (d > 0.f) ? (1.0f / d) : 0.0f;
  }
  __syncthreads();

  // --- epilogue: out[n][c] = acc/deg_n + bias (f32, reference layout)
  float* outb = out + (size_t)b * N_ * C_;
  int mbase = mt * 16 + 8 * h8;
  #pragma unroll
  for (int p = 0; p < 2; ++p) {
    v8f a = p ? acc1 : acc0;
    int c = (ntbase + p) * 16 + cn;
    float bia = bias[c];
    #pragma unroll
    for (int r = 0; r < 8; ++r)
      outb[(size_t)(n0 + mbase + r) * C_ + c] = a[r] * dinv[mbase + r] + bia;
  }
}

// ======================================================================
extern "C" void kernel_launch(void* const* d_in, const int* in_sizes, int n_in,
                              void* d_out, int out_size, void* d_ws, size_t ws_size,
                              hipStream_t stream) {
  const float* x     = (const float*)d_in[0];   // [B,N,Cin]
  const float* H     = (const float*)d_in[1];   // [B,N,E]
  const float* theta = (const float*)d_in[2];   // [Cin,Cout]
  const float* bias  = (const float*)d_in[3];   // [Cout]
  float* out = (float*)d_out;                   // [B,N,Cout] f32

  // workspace: xtT bf16 [B,C,N] (2 MB) | heT bf16 [B,C,E] (1 MB)
  unsigned short* xtT = (unsigned short*)d_ws;
  unsigned short* heT = (unsigned short*)((char*)d_ws + (size_t)B_ * C_ * N_ * 2);

  hg_node_transform<<<B_ * (N_ / 64), 256, 0, stream>>>(x, theta, xtT);
  hg_edge_gemm<<<B_ * (E_ / 64), 256, 0, stream>>>(H, xtT, heT);
  hg_node_gemm<<<B_ * (N_ / 64), 256, 0, stream>>>(H, heT, bias, out);
}
